// SlotAttention_67619965109012
// MI455X (gfx1250) — matrix operand order repaired and checked by reading, not compiled, not run on hardware
//
#include <hip/hip_runtime.h>

#define BB  32
#define NN  4096
#define DD  256
#define HH  512

typedef __bf16 bf16;
typedef __attribute__((ext_vector_type(16))) __bf16 v16bf;
typedef __attribute__((ext_vector_type(8)))  float  v8f;

struct B32b { uint4 a, b; };   // 32 bytes == one bf16 WMMA operand (per lane)

static __device__ __forceinline__ v8f vzero8() {
  v8f z;
  #pragma unroll
  for (int i = 0; i < 8; ++i) z[i] = 0.f;
  return z;
}

static __device__ __forceinline__ unsigned short f2bfbits(float f) {
  unsigned u = __float_as_uint(f);
  u += 0x7fffu + ((u >> 16) & 1u);       // round-to-nearest-even
  return (unsigned short)(u >> 16);
}
static __device__ __forceinline__ bf16 f2bf(float f) {
  unsigned short h = f2bfbits(f);
  return __builtin_bit_cast(bf16, h);
}
static __device__ __forceinline__ float bfbits2f(unsigned h) {
  return __uint_as_float(h << 16);
}

static __device__ __forceinline__ float wsum32(float v) {
  #pragma unroll
  for (int m = 16; m > 0; m >>= 1) v += __shfl_xor(v, m, 32);
  return v;
}

// A operand: rows 0..15 of a row-major bf16 matrix (leading dim ld), K-chunk [k0,k0+32)
// 16-bit A layout: lanes 0-15 hold K = k0+{0..7, 16..23}; lanes 16-31 hold K = k0+{8..15, 24..31}
static __device__ __forceinline__ v16bf load_a(const bf16* A, int ld, int k0, int lane) {
  const int half = (lane >> 4) & 1, li = lane & 15;
  const bf16* p0 = A + (size_t)li * ld + k0 + half * 8;
  B32b t;
  t.a = *(const uint4*)p0;
  t.b = *(const uint4*)(p0 + 16);
  return __builtin_bit_cast(v16bf, t);
}
// B operand from B^T stored row-major [n][k] (leading dim ld):
// lane holds column n = lane&15, K = k0 + (lane<16 ? {0..7,16..23} : {8..15,24..31})
static __device__ __forceinline__ v16bf load_bt(const bf16* BT, int ld, int k0, int lane) {
  const int half = (lane >> 4) & 1, li = lane & 15;
  const bf16* p = BT + (size_t)li * ld + k0 + half * 8;
  B32b t;
  t.a = *(const uint4*)p;
  t.b = *(const uint4*)(p + 16);
  return __builtin_bit_cast(v16bf, t);
}
static __device__ __forceinline__ v8f wmma_bf16(v16bf a, v16bf b, v8f c) {
  return __builtin_amdgcn_wmma_f32_16x16x32_bf16(false, a, false, b, (short)0, c, false, false);
}

// ---------------------------------------------------------------- utilities
__global__ void cvt_bf16_kernel(const float* __restrict__ src, bf16* __restrict__ dst, int n) {
  int i = blockIdx.x * blockDim.x + threadIdx.x;
  if (i < n) dst[i] = f2bf(src[i]);
}

__global__ void slots_init_kernel(const float* __restrict__ mu, float* __restrict__ slots) {
  int i = blockIdx.x * blockDim.x + threadIdx.x;   // B*16*256 elems
  slots[i] = mu[i & (DD - 1)];
}

__global__ void out_kernel(const float* __restrict__ slots, float* __restrict__ out) {
  int idx = blockIdx.x * blockDim.x + threadIdx.x; // B*8*256
  int b = idx >> 11, rem = idx & 2047;
  out[idx] = slots[((size_t)b * 16) * DD + rem];
}

// ------------------------------------------------- fused LN + K/V projection
// grid: (B*N)/32 blocks, 256 threads. Writes k (row-major bf16) and v^T (bf16).
__global__ __launch_bounds__(256) void kv_kernel(
    const float* __restrict__ x, const float* __restrict__ lw, const float* __restrict__ lb,
    const bf16* __restrict__ wk, const bf16* __restrict__ wv,
    const float* __restrict__ bk, const float* __restrict__ bv,
    bf16* __restrict__ kout, bf16* __restrict__ vtout) {
  __shared__ __align__(16) bf16 xs[32][272];
  __shared__ unsigned short vs[32][264];
  const int tid = threadIdx.x, wave = tid >> 5, lane = tid & 31;
  const int half = lane >> 4, li = lane & 15;
  const int m0 = blockIdx.x * 32;          // global row (= b*N + j)
  const int b = m0 >> 12, j0 = m0 & (NN - 1);

  // ---- LayerNorm of 32 input rows into LDS (bf16). Each wave: 4 rows.
  #pragma unroll
  for (int rr = 0; rr < 4; ++rr) {
    const int row = wave * 4 + rr;
    const float* xr = x + (size_t)(m0 + row) * DD + lane * 8;
    float vbuf[8];
    float4 t0 = *(const float4*)xr;
    float4 t1 = *(const float4*)(xr + 4);
    vbuf[0]=t0.x; vbuf[1]=t0.y; vbuf[2]=t0.z; vbuf[3]=t0.w;
    vbuf[4]=t1.x; vbuf[5]=t1.y; vbuf[6]=t1.z; vbuf[7]=t1.w;
    float s = 0.f, s2 = 0.f;
    #pragma unroll
    for (int e = 0; e < 8; ++e) { s += vbuf[e]; s2 += vbuf[e] * vbuf[e]; }
    s = wsum32(s); s2 = wsum32(s2);
    const float mean = s * (1.f / 256.f);
    const float inv = rsqrtf(s2 * (1.f / 256.f) - mean * mean + 1e-5f);
    unsigned pk[4];
    #pragma unroll
    for (int e = 0; e < 8; ++e) {
      const int c = lane * 8 + e;
      float o = (vbuf[e] - mean) * inv * lw[c] + lb[c];
      unsigned short h = f2bfbits(o);
      if (e & 1) pk[e >> 1] |= ((unsigned)h) << 16; else pk[e >> 1] = h;
    }
    *(uint4*)&xs[row][lane * 8] = make_uint4(pk[0], pk[1], pk[2], pk[3]);
  }
  __syncthreads();

  // ---- dual GEMM: wave (g = row group, cw = col group of 64)
  const int g = wave >> 2, cw = wave & 3;
  v8f acck[4], accv[4];
  #pragma unroll
  for (int t = 0; t < 4; ++t) { acck[t] = vzero8(); accv[t] = vzero8(); }
  const bf16* arow = &xs[g * 16][0];
  for (int k0 = 0; k0 < DD; k0 += 32) {
    v16bf a = load_a(arow, 272, k0, lane);
    #pragma unroll
    for (int t = 0; t < 4; ++t) {
      const int c0 = cw * 64 + t * 16;
      acck[t] = wmma_bf16(a, load_bt(wk + (size_t)c0 * DD, DD, k0, lane), acck[t]);
      accv[t] = wmma_bf16(a, load_bt(wv + (size_t)c0 * DD, DD, k0, lane), accv[t]);
    }
  }
  // ---- epilogue: k -> global (bf16, row-major), v -> LDS for transpose
  #pragma unroll
  for (int t = 0; t < 4; ++t) {
    const int col = cw * 64 + t * 16 + li;
    const float bkc = bk[col], bvc = bv[col];
    #pragma unroll
    for (int r = 0; r < 8; ++r) {
      const int lr = g * 16 + r + half * 8;          // block-local row
      kout[(size_t)(m0 + lr) * DD + col] = f2bf(acck[t][r] + bkc);
      vs[lr][col] = f2bfbits(accv[t][r] + bvc);
    }
  }
  __syncthreads();
  // ---- transposed write of v: thread tid owns output dim d = tid
  {
    const int d = tid;
    unsigned wbuf[16];
    #pragma unroll
    for (int jj = 0; jj < 16; ++jj)
      wbuf[jj] = (unsigned)vs[2 * jj][d] | (((unsigned)vs[2 * jj + 1][d]) << 16);
    uint4* dst = (uint4*)(vtout + ((size_t)b * DD + d) * NN + j0);
    dst[0] = make_uint4(wbuf[0], wbuf[1], wbuf[2], wbuf[3]);
    dst[1] = make_uint4(wbuf[4], wbuf[5], wbuf[6], wbuf[7]);
    dst[2] = make_uint4(wbuf[8], wbuf[9], wbuf[10], wbuf[11]);
    dst[3] = make_uint4(wbuf[12], wbuf[13], wbuf[14], wbuf[15]);
  }
}

// ------------------------------------------------- q = LN(slots)@Wq^T + bq (x SCALE)
// grid: B blocks. q stored bf16 [b][16][256], SCALE folded in.
__global__ __launch_bounds__(256) void q_kernel(
    const float* __restrict__ slots, const float* __restrict__ lw, const float* __restrict__ lb,
    const bf16* __restrict__ wq, const float* __restrict__ bq, bf16* __restrict__ qout) {
  __shared__ __align__(16) bf16 xs[16][272];
  const int tid = threadIdx.x, wave = tid >> 5, lane = tid & 31;
  const int half = lane >> 4, li = lane & 15, b = blockIdx.x;
  #pragma unroll
  for (int rr = 0; rr < 2; ++rr) {
    const int row = wave * 2 + rr;
    const float* xr = slots + ((size_t)b * 16 + row) * DD + lane * 8;
    float vbuf[8];
    float4 t0 = *(const float4*)xr;
    float4 t1 = *(const float4*)(xr + 4);
    vbuf[0]=t0.x; vbuf[1]=t0.y; vbuf[2]=t0.z; vbuf[3]=t0.w;
    vbuf[4]=t1.x; vbuf[5]=t1.y; vbuf[6]=t1.z; vbuf[7]=t1.w;
    float s = 0.f, s2 = 0.f;
    #pragma unroll
    for (int e = 0; e < 8; ++e) { s += vbuf[e]; s2 += vbuf[e] * vbuf[e]; }
    s = wsum32(s); s2 = wsum32(s2);
    const float mean = s * (1.f / 256.f);
    const float inv = rsqrtf(s2 * (1.f / 256.f) - mean * mean + 1e-5f);
    unsigned pk[4];
    #pragma unroll
    for (int e = 0; e < 8; ++e) {
      const int c = lane * 8 + e;
      float o = (vbuf[e] - mean) * inv * lw[c] + lb[c];
      unsigned short h = f2bfbits(o);
      if (e & 1) pk[e >> 1] |= ((unsigned)h) << 16; else pk[e >> 1] = h;
    }
    *(uint4*)&xs[row][lane * 8] = make_uint4(pk[0], pk[1], pk[2], pk[3]);
  }
  __syncthreads();
  v8f acc[2] = {vzero8(), vzero8()};
  for (int k0 = 0; k0 < DD; k0 += 32) {
    v16bf a = load_a(&xs[0][0], 272, k0, lane);
    #pragma unroll
    for (int t = 0; t < 2; ++t) {
      const int c0 = (wave * 2 + t) * 16;
      acc[t] = wmma_bf16(a, load_bt(wq + (size_t)c0 * DD, DD, k0, lane), acc[t]);
    }
  }
  #pragma unroll
  for (int t = 0; t < 2; ++t) {
    const int col = (wave * 2 + t) * 16 + li;
    const float bias = bq[col];
    #pragma unroll
    for (int r = 0; r < 8; ++r) {
      const int row = r + half * 8;
      qout[((size_t)b * 16 + row) * DD + col] = f2bf((acc[t][r] + bias) * 0.0625f); // D^-0.5
    }
  }
}

// ------------------------------------------------- dots^T = k . q^T  + softmax over slots
// grid: B * (N/128), 8 waves each own 16 j-rows. attn bf16 [b][16][N], rows 8..15 = 0.
__global__ __launch_bounds__(256) void dots_kernel(
    const bf16* __restrict__ kmat, const bf16* __restrict__ qmat, bf16* __restrict__ attn) {
  const int tid = threadIdx.x, wave = tid >> 5, lane = tid & 31;
  const int half = lane >> 4, li = lane & 15;
  const int blk = blockIdx.x, b = blk >> 5;
  const int j0 = (blk & 31) * 128 + wave * 16;
  const bf16* arow = kmat + ((size_t)b * NN + j0) * DD;
  const bf16* bt = qmat + (size_t)b * 16 * DD;
  v8f acc = vzero8();
  for (int k0 = 0; k0 < DD; k0 += 32)
    acc = wmma_bf16(load_a(arow, DD, k0, lane), load_bt(bt, DD, k0, lane), acc);
  // acc[r] = dots for j = j0 + r + half*8, slot i = li (valid i<8). Softmax across lanes (width 8).
  unsigned pk[4];
  #pragma unroll
  for (int r = 0; r < 8; ++r) {
    float d = acc[r];
    float m = d;
    #pragma unroll
    for (int mm = 1; mm < 8; mm <<= 1) m = fmaxf(m, __shfl_xor(m, mm, 8));
    float e = __expf(d - m);
    float s = e;
    #pragma unroll
    for (int mm = 1; mm < 8; mm <<= 1) s += __shfl_xor(s, mm, 8);
    float a = (li < 8) ? (e / s + 1e-8f) : 0.f;   // pad slots store zeros
    unsigned short h = f2bfbits(a);
    if (r & 1) pk[r >> 1] |= ((unsigned)h) << 16; else pk[r >> 1] = h;
  }
  bf16* dst = attn + ((size_t)b * 16 + li) * NN + j0 + half * 8;
  *(uint4*)dst = make_uint4(pk[0], pk[1], pk[2], pk[3]);
}

// ------------------------------------------------- S[b][i] = sum_j attn[b][i][j]
// grid: B*8 blocks (one slot row each); each thread loads exactly 32B.
__global__ __launch_bounds__(256) void s_kernel(const bf16* __restrict__ attn, float* __restrict__ S) {
  __shared__ float red[8];
  const int blk = blockIdx.x, b = blk >> 3, i = blk & 7;
  const int tid = threadIdx.x, wave = tid >> 5, lane = tid & 31;
  const bf16* row = attn + ((size_t)b * 16 + i) * NN;
  const uint4 u0 = *(const uint4*)(row + tid * 16);
  const uint4 u1 = *(const uint4*)(row + tid * 16 + 8);
  const unsigned w[8] = {u0.x, u0.y, u0.z, u0.w, u1.x, u1.y, u1.z, u1.w};
  float s = 0.f;
  #pragma unroll
  for (int q = 0; q < 8; ++q)
    s += bfbits2f(w[q] & 0xffffu) + bfbits2f(w[q] >> 16);
  s = wsum32(s);
  if (lane == 0) red[wave] = s;
  __syncthreads();
  if (tid == 0) {
    float t = 0.f;
    #pragma unroll
    for (int ww = 0; ww < 8; ++ww) t += red[ww];
    S[b * 16 + i] = fmaxf(t, 1e-30f);
    S[b * 16 + i + 8] = 1.0f;               // pad rows: safe divisor
  }
}

// ------------------------------------------------- upd = (attn @ v) / S   (fp32 out)
// grid: B*16 blocks (one 16-col d-tile each); 8 waves split the j-range; LDS tile reduction.
__global__ __launch_bounds__(256) void upd_kernel(
    const bf16* __restrict__ attn, const bf16* __restrict__ vt,
    const float* __restrict__ S, float* __restrict__ upd) {
  __shared__ float part[8][16][16];
  const int tid = threadIdx.x, wave = tid >> 5, lane = tid & 31;
  const int half = lane >> 4, li = lane & 15;
  const int blk = blockIdx.x, b = blk >> 4, c0 = (blk & 15) * 16;
  const bf16* arow = attn + (size_t)b * 16 * NN;
  const bf16* btrow = vt + ((size_t)b * DD + c0) * NN;
  v8f acc = vzero8();
  const int kbeg = wave * (NN / 8), kend = kbeg + (NN / 8);
  for (int k0 = kbeg; k0 < kend; k0 += 128) {
    // software prefetch of the next 256B chunk of both streams (global_prefetch_b8)
    __builtin_prefetch(arow + (size_t)li * NN + k0 + 128, 0, 1);
    __builtin_prefetch(btrow + (size_t)li * NN + k0 + 128, 0, 1);
    #pragma unroll
    for (int kk = 0; kk < 128; kk += 32)
      acc = wmma_bf16(load_a(arow, NN, k0 + kk, lane),
                      load_bt(btrow, NN, k0 + kk, lane), acc);
  }
  #pragma unroll
  for (int r = 0; r < 8; ++r) part[wave][r + half * 8][li] = acc[r];
  __syncthreads();
  const int row = tid >> 4, col = tid & 15;
  float s = 0.f;
  #pragma unroll
  for (int w = 0; w < 8; ++w) s += part[w][row][col];
  upd[((size_t)b * 16 + row) * DD + c0 + col] = s / S[b * 16 + row];
}

// ------------------------------------------------- gi = upd@Wih^T + bih ; gh = slots@Whh^T + bhh
// grid: B blocks; wave owns 6 col-tiles (96 cols of 768) for both GEMMs.
__global__ __launch_bounds__(256) void gigh_kernel(
    const float* __restrict__ upd, const float* __restrict__ slots,
    const bf16* __restrict__ wih, const bf16* __restrict__ whh,
    const float* __restrict__ bih, const float* __restrict__ bhh,
    float* __restrict__ gi, float* __restrict__ gh) {
  __shared__ __align__(16) bf16 au[16][272];
  __shared__ __align__(16) bf16 as[16][272];
  const int b = blockIdx.x, tid = threadIdx.x, wave = tid >> 5, lane = tid & 31;
  const int half = lane >> 4, li = lane & 15;
  {
    const int row = tid >> 4, c = (tid & 15) * 16;
    const float* su = upd + ((size_t)b * 16 + row) * DD + c;
    const float* ss = slots + ((size_t)b * 16 + row) * DD + c;
    #pragma unroll
    for (int e = 0; e < 16; ++e) { au[row][c + e] = f2bf(su[e]); as[row][c + e] = f2bf(ss[e]); }
  }
  __syncthreads();
  for (int t = 0; t < 6; ++t) {
    const int c0 = wave * 96 + t * 16;
    v8f ai = vzero8(), ah = vzero8();
    for (int k0 = 0; k0 < DD; k0 += 32) {
      v16bf aU = load_a(&au[0][0], 272, k0, lane);
      v16bf aS = load_a(&as[0][0], 272, k0, lane);
      ai = wmma_bf16(aU, load_bt(wih + (size_t)c0 * DD, DD, k0, lane), ai);
      ah = wmma_bf16(aS, load_bt(whh + (size_t)c0 * DD, DD, k0, lane), ah);
    }
    const int col = c0 + li;
    const float bi = bih[col], bh = bhh[col];
    #pragma unroll
    for (int r = 0; r < 8; ++r) {
      const int i = r + half * 8;
      gi[((size_t)b * 16 + i) * 768 + col] = ai[r] + bi;
      gh[((size_t)b * 16 + i) * 768 + col] = ah[r] + bh;
    }
  }
}

// ------------------------------------------------- GRU gates + LN_ff. grid: B*8 (one valid slot row each)
__global__ __launch_bounds__(256) void gate_kernel(
    const float* __restrict__ gi, const float* __restrict__ gh, const float* __restrict__ slots,
    const float* __restrict__ lw, const float* __restrict__ lb,
    float* __restrict__ slots2, float* __restrict__ xff) {
  const int blk = blockIdx.x, b = blk >> 3, i = blk & 7, d = threadIdx.x;
  const size_t base = (size_t)b * 16 + i;
  const float ir = gi[base * 768 + d], iz = gi[base * 768 + 256 + d], in_ = gi[base * 768 + 512 + d];
  const float hr = gh[base * 768 + d], hz = gh[base * 768 + 256 + d], hn = gh[base * 768 + 512 + d];
  const float r = 1.f / (1.f + __expf(-(ir + hr)));
  const float z = 1.f / (1.f + __expf(-(iz + hz)));
  const float n = tanhf(in_ + r * hn);
  const float sn = (1.f - z) * n + z * slots[base * DD + d];
  slots2[base * DD + d] = sn;
  __shared__ float red[16];
  float s = wsum32(sn), s2 = wsum32(sn * sn);
  const int wave = d >> 5, lane = d & 31;
  if (lane == 0) { red[wave] = s; red[8 + wave] = s2; }
  __syncthreads();
  float ts = 0.f, ts2 = 0.f;
  #pragma unroll
  for (int w = 0; w < 8; ++w) { ts += red[w]; ts2 += red[8 + w]; }
  const float mean = ts * (1.f / 256.f);
  const float inv = rsqrtf(ts2 * (1.f / 256.f) - mean * mean + 1e-5f);
  xff[base * DD + d] = (sn - mean) * inv * lw[d] + lb[d];
}

// ------------------------------------------------- residual MLP: slots = slots2 + relu(xff@W1^T+b1)@W2^T + b2
__global__ __launch_bounds__(256) void mlp_kernel(
    const float* __restrict__ xff, const float* __restrict__ slots2,
    const bf16* __restrict__ w1, const float* __restrict__ b1,
    const bf16* __restrict__ w2, const float* __restrict__ b2,
    float* __restrict__ slots) {
  __shared__ __align__(16) bf16 xs[16][272];
  __shared__ __align__(16) bf16 hs[16][528];
  const int b = blockIdx.x, tid = threadIdx.x, wave = tid >> 5, lane = tid & 31;
  const int half = lane >> 4, li = lane & 15;
  {
    const int row = tid >> 4, c = (tid & 15) * 16;
    const float* s = xff + ((size_t)b * 16 + row) * DD + c;
    #pragma unroll
    for (int e = 0; e < 16; ++e) xs[row][c + e] = f2bf(s[e]);
  }
  __syncthreads();
  // GEMM1: [16x256] x [256x512] -> relu -> hs
  for (int t = 0; t < 4; ++t) {
    const int c0 = wave * 64 + t * 16;
    v8f acc = vzero8();
    for (int k0 = 0; k0 < DD; k0 += 32)
      acc = wmma_bf16(load_a(&xs[0][0], 272, k0, lane),
                      load_bt(w1 + (size_t)c0 * DD, DD, k0, lane), acc);
    const int col = c0 + li;
    const float bias = b1[col];
    #pragma unroll
    for (int r = 0; r < 8; ++r) {
      const int i = r + half * 8;
      hs[i][col] = f2bf(fmaxf(acc[r] + bias, 0.f));
    }
  }
  __syncthreads();
  // GEMM2: [16x512] x [512x256] + residual
  for (int t = 0; t < 2; ++t) {
    const int c0 = wave * 32 + t * 16;
    v8f acc = vzero8();
    for (int k0 = 0; k0 < HH; k0 += 32)
      acc = wmma_bf16(load_a(&hs[0][0], 528, k0, lane),
                      load_bt(w2 + (size_t)c0 * HH, HH, k0, lane), acc);
    const int col = c0 + li;
    const float bias = b2[col];
    #pragma unroll
    for (int r = 0; r < 8; ++r) {
      const int i = r + half * 8;
      const size_t o = ((size_t)b * 16 + i) * DD + col;
      slots[o] = acc[r] + bias + slots2[o];
    }
  }
}

// ==================================================================== host
extern "C" void kernel_launch(void* const* d_in, const int* in_sizes, int n_in,
                              void* d_out, int out_size, void* d_ws, size_t ws_size,
                              hipStream_t stream) {
  (void)in_sizes; (void)n_in; (void)out_size; (void)ws_size;
  const float* inputs    = (const float*)d_in[0];
  const float* slots_mu  = (const float*)d_in[1];
  const float* ln_in_w   = (const float*)d_in[2];
  const float* ln_in_b   = (const float*)d_in[3];
  const float* Wq        = (const float*)d_in[4];
  const float* bq        = (const float*)d_in[5];
  const float* Wk        = (const float*)d_in[6];
  const float* bk        = (const float*)d_in[7];
  const float* Wv        = (const float*)d_in[8];
  const float* bv        = (const float*)d_in[9];
  const float* ln_sl_w   = (const float*)d_in[10];
  const float* ln_sl_b   = (const float*)d_in[11];
  const float* W_ih      = (const float*)d_in[12];
  const float* b_ih      = (const float*)d_in[13];
  const float* W_hh      = (const float*)d_in[14];
  const float* b_hh      = (const float*)d_in[15];
  const float* ln_ff_w   = (const float*)d_in[16];
  const float* ln_ff_b   = (const float*)d_in[17];
  const float* W1        = (const float*)d_in[18];
  const float* b1        = (const float*)d_in[19];
  const float* W2        = (const float*)d_in[20];
  const float* b2        = (const float*)d_in[21];

  char* cur = (char*)d_ws;
  auto alloc = [&](size_t bytes) -> void* {
    void* p = (void*)cur;
    cur += (bytes + 255) & ~(size_t)255;
    return p;
  };
  bf16* wq_bf  = (bf16*)alloc((size_t)DD * DD * 2);
  bf16* wk_bf  = (bf16*)alloc((size_t)DD * DD * 2);
  bf16* wv_bf  = (bf16*)alloc((size_t)DD * DD * 2);
  bf16* wih_bf = (bf16*)alloc((size_t)3 * DD * DD * 2);
  bf16* whh_bf = (bf16*)alloc((size_t)3 * DD * DD * 2);
  bf16* w1_bf  = (bf16*)alloc((size_t)HH * DD * 2);
  bf16* w2_bf  = (bf16*)alloc((size_t)DD * HH * 2);
  bf16* k_bf   = (bf16*)alloc((size_t)BB * NN * DD * 2);
  bf16* vt_bf  = (bf16*)alloc((size_t)BB * NN * DD * 2);
  bf16* q_bf   = (bf16*)alloc((size_t)BB * 16 * DD * 2);
  bf16* att_bf = (bf16*)alloc((size_t)BB * 16 * NN * 2);
  float* Sbuf  = (float*)alloc((size_t)BB * 16 * 4);
  float* upd   = (float*)alloc((size_t)BB * 16 * DD * 4);
  float* slots = (float*)alloc((size_t)BB * 16 * DD * 4);
  float* slots2= (float*)alloc((size_t)BB * 16 * DD * 4);
  float* xff   = (float*)alloc((size_t)BB * 16 * DD * 4);
  float* gi    = (float*)alloc((size_t)BB * 16 * 768 * 4);
  float* gh    = (float*)alloc((size_t)BB * 16 * 768 * 4);

  auto cvt = [&](const float* s, bf16* d, int n) {
    cvt_bf16_kernel<<<(n + 255) / 256, 256, 0, stream>>>(s, d, n);
  };
  cvt(Wq, wq_bf, DD * DD);
  cvt(Wk, wk_bf, DD * DD);
  cvt(Wv, wv_bf, DD * DD);
  cvt(W_ih, wih_bf, 3 * DD * DD);
  cvt(W_hh, whh_bf, 3 * DD * DD);
  cvt(W1, w1_bf, HH * DD);
  cvt(W2, w2_bf, DD * HH);

  slots_init_kernel<<<(BB * 16 * DD) / 256, 256, 0, stream>>>(slots_mu, slots);
  kv_kernel<<<(BB * NN) / 32, 256, 0, stream>>>(inputs, ln_in_w, ln_in_b,
                                                wk_bf, wv_bf, bk, bv, k_bf, vt_bf);
  for (int it = 0; it < 3; ++it) {
    q_kernel<<<BB, 256, 0, stream>>>(slots, ln_sl_w, ln_sl_b, wq_bf, bq, q_bf);
    dots_kernel<<<BB * 32, 256, 0, stream>>>(k_bf, q_bf, att_bf);
    s_kernel<<<BB * 8, 256, 0, stream>>>(att_bf, Sbuf);
    upd_kernel<<<BB * 16, 256, 0, stream>>>(att_bf, vt_bf, Sbuf, upd);
    gigh_kernel<<<BB, 256, 0, stream>>>(upd, slots, wih_bf, whh_bf, b_ih, b_hh, gi, gh);
    gate_kernel<<<BB * 8, 256, 0, stream>>>(gi, gh, slots, ln_ff_w, ln_ff_b, slots2, xff);
    mlp_kernel<<<BB, 256, 0, stream>>>(xff, slots2, w1_bf, b1, w2_bf, b2, slots);
  }
  out_kernel<<<(BB * 8 * DD) / 256, 256, 0, stream>>>(slots, (float*)d_out);
}
